// SimpleGNN_13580686590510
// MI455X (gfx1250) — compile-verified, matrix operand order checked by reading, not run on hardware
//
#include <hip/hip_runtime.h>
#include <hip/hip_bf16.h>

// ---------------------------------------------------------------------------
// Types for CDNA5 WMMA (wave32): bf16 A/B fragments -> v16bf, f32 C/D -> v8f.
// ---------------------------------------------------------------------------
typedef __bf16 bf16_t;
typedef __attribute__((ext_vector_type(4)))  __bf16 v4bf;
typedef __attribute__((ext_vector_type(8)))  __bf16 v8bf;
typedef __attribute__((ext_vector_type(16))) __bf16 v16bf;
typedef __attribute__((ext_vector_type(8)))  float  v8f;

static __device__ inline v16bf cat16(v8bf lo, v8bf hi) {
  return __builtin_shufflevector(lo, hi, 0, 1, 2, 3, 4, 5, 6, 7,
                                         8, 9, 10, 11, 12, 13, 14, 15);
}

#define DFEAT 128
#define LDS_STRIDE 136           // 128 + 8 halves padding
#define LDS_ROW_BYTES (LDS_STRIDE * 2)  // 272 B, 16 B aligned

// ---------------------------------------------------------------------------
// Zero a float buffer with 16 B stores.
// ---------------------------------------------------------------------------
__global__ void gnn_zero_kernel(float4* __restrict__ p, int n4) {
  int i = blockIdx.x * blockDim.x + threadIdx.x;
  if (i < n4) p[i] = make_float4(0.f, 0.f, 0.f, 0.f);
}

// ---------------------------------------------------------------------------
// SpMM scatter: out[row[e]] += val[e] * x[col[e]].
// One wave32 per edge; lane owns a float4 (16 B) of the 512 B feature row.
// Atomic f32 adds land in the 192 MB L2 (dest buffer is 51.2 MB).
// ---------------------------------------------------------------------------
__global__ __launch_bounds__(256) void gnn_spmm_kernel(
    const int* __restrict__ erow, const int* __restrict__ ecol,
    const float* __restrict__ eval_, const float* __restrict__ x,
    float* __restrict__ out, int E) {
  int e = blockIdx.x * 8 + (threadIdx.x >> 5);
  if (e >= E) return;
  int lane = threadIdx.x & 31;
  int r = erow[e];
  int c = ecol[e];
  float v = eval_[e];
  float4 d = *(const float4*)(x + (size_t)c * DFEAT + lane * 4);
  float* o = out + (size_t)r * DFEAT + lane * 4;
  unsafeAtomicAdd(o + 0, v * d.x);   // global_atomic_add_f32 (no-return)
  unsafeAtomicAdd(o + 1, v * d.y);
  unsafeAtomicAdd(o + 2, v * d.z);
  unsafeAtomicAdd(o + 3, v * d.w);
}

// ---------------------------------------------------------------------------
// Elementwise f32 -> bf16 convert (activations): 4 floats -> one 8 B store.
// ---------------------------------------------------------------------------
__global__ void gnn_cvt_kernel(const float4* __restrict__ in,
                               bf16_t* __restrict__ out, int n4) {
  int i = blockIdx.x * blockDim.x + threadIdx.x;
  if (i >= n4) return;
  float4 f = in[i];
  v4bf o = {(__bf16)f.x, (__bf16)f.y, (__bf16)f.z, (__bf16)f.w};
  *(v4bf*)(out + (size_t)i * 4) = o;
}

// ---------------------------------------------------------------------------
// Weight convert + transpose: Wt[n][k] = bf16(W[k][n]), 128x128.
// ---------------------------------------------------------------------------
__global__ void gnn_wcvt_kernel(const float* __restrict__ W,
                                bf16_t* __restrict__ Wt) {
  int i = blockIdx.x * blockDim.x + threadIdx.x;  // 16384 total
  int k = i >> 7, n = i & 127;
  Wt[(size_t)n * DFEAT + k] = (bf16_t)W[(size_t)k * DFEAT + n];
}

// ---------------------------------------------------------------------------
// Dense layer: out = relu(A @ W + b). A given as bf16 [NP,128] row-major,
// Wt as bf16 [128][128] pre-transposed (Wt[n][k]).
// 256 threads (8 waves), block computes a full 128-row strip (NP padded to a
// multiple of 128, so no bounds checks anywhere).
// Staging uses gfx1250 async copies: global_load_async_to_lds_b128 with
// per-lane LDS destinations (keeps the 272 B padded row stride), then
// s_wait_asynccnt 0 + workgroup barrier.
// Each wave: 16 rows x 128 cols, K=128 in 4 steps of 32
//   -> 32x v_wmma_f32_16x16x32_bf16, fragments via paired ds_load_b128
//      (ISA 7.12.2: element i of v16bf <-> K = 16*(i>>3) + 8*(lane>>4) + (i&7)).
// ---------------------------------------------------------------------------
__global__ __launch_bounds__(256, 1) void gnn_gemm_relu_kernel(
    const bf16_t* __restrict__ Abf, const bf16_t* __restrict__ Wt,
    const float* __restrict__ b, float* __restrict__ out) {
  __shared__ bf16_t la[DFEAT][LDS_STRIDE];
  __shared__ bf16_t lw[DFEAT][LDS_STRIDE];

  const int tid = threadIdx.x;
  const int m0 = blockIdx.x * 128;

  const unsigned laBase = (unsigned)(size_t)&la[0][0];
  const unsigned lwBase = (unsigned)(size_t)&lw[0][0];
  const bf16_t* gA = Abf + (size_t)m0 * DFEAT;

  // Async stage A tile: 128 rows x 16 chunks of 16 B (8 bf16) each.
  for (int idx = tid; idx < 128 * 16; idx += 256) {
    int r = idx >> 4, ch = idx & 15;
    const void* gp = gA + r * DFEAT + ch * 8;
    unsigned lp = laBase + r * LDS_ROW_BYTES + ch * 16;
    asm volatile("global_load_async_to_lds_b128 %0, %1, off"
                 :: "v"(lp), "v"(gp) : "memory");
  }
  // Async stage Wt tile (128x128 bf16).
  for (int idx = tid; idx < 128 * 16; idx += 256) {
    int r = idx >> 4, ch = idx & 15;
    const void* gp = Wt + r * DFEAT + ch * 8;
    unsigned lp = lwBase + r * LDS_ROW_BYTES + ch * 16;
    asm volatile("global_load_async_to_lds_b128 %0, %1, off"
                 :: "v"(lp), "v"(gp) : "memory");
  }
  asm volatile("s_wait_asynccnt 0" ::: "memory");
  __syncthreads();

  const int wave = tid >> 5;
  const int lane = tid & 31;
  const int g = lane >> 4;     // K-half select
  const int nl = lane & 15;    // A row / B col within tile

  // A fragments for this wave's 16 rows, all 4 K-blocks.
  v16bf afrag[4];
#pragma unroll
  for (int kb = 0; kb < 4; ++kb) {
    const bf16_t* ap = &la[wave * 16 + nl][kb * 32 + 8 * g];
    v8bf lo = *(const v8bf*)ap;          // K = kb*32 + 8g .. +7
    v8bf hi = *(const v8bf*)(ap + 16);   // K = kb*32 + 16 + 8g .. +7
    afrag[kb] = cat16(lo, hi);
  }

#pragma unroll
  for (int t = 0; t < 8; ++t) {
    const int ncol = t * 16 + nl;
    const float bias = b[ncol];
    v8f c = {bias, bias, bias, bias, bias, bias, bias, bias};
#pragma unroll
    for (int kb = 0; kb < 4; ++kb) {
      const bf16_t* wp = &lw[ncol][kb * 32 + 8 * g];
      v8bf lo = *(const v8bf*)wp;
      v8bf hi = *(const v8bf*)(wp + 16);
      v16bf bfrag = cat16(lo, hi);
      c = __builtin_amdgcn_wmma_f32_16x16x32_bf16(
          /*neg_a=*/false, afrag[kb], /*neg_b=*/false, bfrag,
          /*c_mod=*/(short)0, c, /*reuse_a=*/false, /*reuse_b=*/false);
    }
    // C/D layout: VGPR j -> M = j + 8*g; lane nl -> N = ncol. No guards (padded).
    float* orow = out + (size_t)(m0 + wave * 16 + 8 * g) * DFEAT + ncol;
#pragma unroll
    for (int j = 0; j < 8; ++j) {
      orow[(size_t)j * DFEAT] = fmaxf(c[j], 0.0f);
    }
  }
}

// ---------------------------------------------------------------------------
// Output head: out[r] = softplus(h[r] . Wout + bout). One wave32 per row.
// ---------------------------------------------------------------------------
__global__ __launch_bounds__(256) void gnn_head_kernel(
    const float* __restrict__ h, const float* __restrict__ Wout,
    const float* __restrict__ bout, float* __restrict__ out, int N) {
  int row = blockIdx.x * 8 + (threadIdx.x >> 5);
  if (row >= N) return;
  int lane = threadIdx.x & 31;
  float4 hv = *(const float4*)(h + (size_t)row * DFEAT + lane * 4);
  float4 wv = *(const float4*)(Wout + lane * 4);
  float s = hv.x * wv.x + hv.y * wv.y + hv.z * wv.z + hv.w * wv.w;
  s += __shfl_xor(s, 16, 32);
  s += __shfl_xor(s, 8, 32);
  s += __shfl_xor(s, 4, 32);
  s += __shfl_xor(s, 2, 32);
  s += __shfl_xor(s, 1, 32);
  if (lane == 0) {
    float z = s + bout[0];
    out[row] = (z > 20.0f) ? z : log1pf(expf(z));
  }
}

// ---------------------------------------------------------------------------
// Launch. NP = N padded to a multiple of 128 rows (kills all GEMM guards).
// ws layout: bufA f32[NP*128] | bufB f32[NP*128] | bufAbf bf16[NP*128] |
//            wtbf bf16[128*128]
// ---------------------------------------------------------------------------
extern "C" void kernel_launch(void* const* d_in, const int* in_sizes, int n_in,
                              void* d_out, int out_size, void* d_ws, size_t ws_size,
                              hipStream_t stream) {
  const float* x    = (const float*)d_in[0];
  const int*   erow = (const int*)d_in[1];
  const int*   ecol = (const int*)d_in[2];
  const float* ev   = (const float*)d_in[3];
  const float* W1   = (const float*)d_in[4];
  const float* b1   = (const float*)d_in[5];
  const float* W2   = (const float*)d_in[6];
  const float* b2   = (const float*)d_in[7];
  const float* Wout = (const float*)d_in[8];
  const float* bout = (const float*)d_in[9];
  float* out = (float*)d_out;

  const int N  = in_sizes[0] / DFEAT;
  const int E  = in_sizes[1];
  const int NP = ((N + 127) / 128) * 128;

  float*  bufA   = (float*)d_ws;
  float*  bufB   = bufA + (size_t)NP * DFEAT;
  bf16_t* bufAbf = (bf16_t*)(bufB + (size_t)NP * DFEAT);
  bf16_t* wtbf   = bufAbf + (size_t)NP * DFEAT;

  const int n4 = (NP * DFEAT) / 4;
  dim3 blk(256);
  dim3 zgrid((n4 + 255) / 256);
  dim3 sgrid((E + 7) / 8);
  dim3 ggrid(NP / 128);
  dim3 wgrid((DFEAT * DFEAT) / 256);
  dim3 hgrid((N + 7) / 8);

  // Layer 1
  gnn_zero_kernel<<<zgrid, blk, 0, stream>>>((float4*)bufA, n4);
  gnn_spmm_kernel<<<sgrid, blk, 0, stream>>>(erow, ecol, ev, x, bufA, E);
  gnn_cvt_kernel<<<zgrid, blk, 0, stream>>>((const float4*)bufA, bufAbf, n4);
  gnn_wcvt_kernel<<<wgrid, blk, 0, stream>>>(W1, wtbf);
  gnn_gemm_relu_kernel<<<ggrid, blk, 0, stream>>>(bufAbf, wtbf, b1, bufB);
  // Layer 2
  gnn_zero_kernel<<<zgrid, blk, 0, stream>>>((float4*)bufA, n4);
  gnn_spmm_kernel<<<sgrid, blk, 0, stream>>>(erow, ecol, ev, bufB, bufA, E);
  gnn_cvt_kernel<<<zgrid, blk, 0, stream>>>((const float4*)bufA, bufAbf, n4);
  gnn_wcvt_kernel<<<wgrid, blk, 0, stream>>>(W2, wtbf);
  gnn_gemm_relu_kernel<<<ggrid, blk, 0, stream>>>(bufAbf, wtbf, b2, bufB);
  // Final aggregation + head
  gnn_zero_kernel<<<zgrid, blk, 0, stream>>>((float4*)bufA, n4);
  gnn_spmm_kernel<<<sgrid, blk, 0, stream>>>(erow, ecol, ev, bufB, bufA, E);
  gnn_head_kernel<<<hgrid, blk, 0, stream>>>(bufA, Wout, bout, out, N);
}
